// GPTLanguageModel_16947940950813
// MI455X (gfx1250) — matrix operand order repaired and checked by reading, not compiled, not run on hardware
//
#include <hip/hip_runtime.h>

// ---------------------------------------------------------------------------
// GPT block forward for MI455X (gfx1250, wave32, WMMA + Tensor Data Mover).
// GEMM k-tiles are DMA'd into double-buffered LDS by the TDM
// (tensor_load_to_lds) while the 8 waves run v_wmma_f32_16x16x32_f16.
// Workspace requirement: ~483 MB (layout below).
// ---------------------------------------------------------------------------

typedef _Float16 h16;
typedef __attribute__((ext_vector_type(16))) _Float16 v16h;
typedef __attribute__((ext_vector_type(8)))  _Float16 v8h;
typedef __attribute__((ext_vector_type(8)))  float    v8f;
typedef __attribute__((ext_vector_type(4)))  unsigned int u32x4;
typedef __attribute__((ext_vector_type(8)))  int      i32x8;

#define NTHREADS 256
#define BM 128
#define BN 128
#define BK 32
#define SA 40   // LDS row stride (halves): 32 data + 8 pad -> 16dw + 4dw pad
#define SB 40

// ---------------------------------------------------------------------------
// Issue one TDM 2D tile load: BMxBK f16 tile, row stride `ld` elements,
// into LDS at byte offset lds_off with the SA/SB padding applied by the TDM.
// D# layout per CDNA5 ISA (groups 0 and 1, 2D tensor).
// Must be executed by one wave (values are wave-uniform; EXEC is ignored).
// ---------------------------------------------------------------------------
__device__ __forceinline__ void tdm_load_tile_f16(unsigned int lds_off,
                                                  const h16* gsrc,
                                                  long long ld)
{
    const unsigned long long ga = (unsigned long long)(uintptr_t)gsrc;
    u32x4 g0;
    g0[0] = 1u;                                        // count=1 (valid), user mode
    g0[1] = lds_off;                                   // lds_addr (bytes)
    g0[2] = (unsigned int)ga;                          // global_addr[31:0]
    g0[3] = (unsigned int)(ga >> 32) | (2u << 30);     // global_addr[56:32] | type=2

    i32x8 g1;
    g1[0] = (int)((1u << 16)      // data_size = 2 bytes
                | (1u << 20)      // pad_enable
                | (3u << 22)      // pad_interval: 16 dwords (=64B tile row)
                | (3u << 25));    // pad_amount  :  4 dwords (=8 halves pad)
    const unsigned int td = 1u << 20;                  // oversized dims: never clip
    g1[1] = (int)((td & 0xFFFFu) << 16);               // tensor_dim0[15:0]
    g1[2] = (int)((td >> 16) | ((td & 0xFFFFu) << 16));// tensor_dim0[31:16] | dim1 lo
    g1[3] = (int)((td >> 16) | ((unsigned)BK << 16));  // tensor_dim1[31:16] | tile_dim0=32
    g1[4] = BM;                                        // tile_dim1=128, tile_dim2=0
    const unsigned long long st = (unsigned long long)ld;
    g1[5] = (int)(unsigned int)st;                     // tensor_dim0_stride[31:0]
    g1[6] = (int)((unsigned int)(st >> 32) & 0xFFFFu); // stride[47:32]; dim1_stride=0
    g1[7] = 0;

    asm volatile("tensor_load_to_lds %0, %1" :: "s"(g0), "s"(g1) : "memory");
}

// ---------------------------------------------------------------------------
// Generic WMMA GEMM:  C[z][m][n] = sum_k A[z][m][k] * Bt[z][n][k]  (+bias[n])
//                     (+residual) (ReLU) -> f32 out and/or f16 out
// 256 threads = 8 waves, arranged 4(M) x 2(N); wave tile 32x64; block 128x128.
// Wave0 DMAs A tiles, wave1 DMAs B tiles, one k-step ahead (double buffer).
// causal=1 skips blocks strictly above the diagonal (n0 > m0+BM-1).
// ---------------------------------------------------------------------------
__global__ __launch_bounds__(NTHREADS)
void gemm_wmma_f16(const h16* __restrict__ A,  int lda, long long zA,
                   const h16* __restrict__ Bt, int ldb, long long zB,
                   float*     __restrict__ Cf, int ldcf, long long zCf,
                   h16*       __restrict__ Ch, int ldch, long long zCh,
                   const float* __restrict__ bias,
                   const float* __restrict__ residual,  // layout of Cf
                   int K, int relu, int causal)
{
    const int m0 = blockIdx.y * BM;
    const int n0 = blockIdx.x * BN;
    if (causal && n0 > m0 + (BM - 1)) return;   // skip upper-triangular blocks

    const long long z = blockIdx.z;
    A  += z * zA + (long long)m0 * lda;         // A tile origin (k=0)
    Bt += z * zB + (long long)n0 * ldb;         // B tile origin (k=0)

    __shared__ h16 As[2][BM * SA];
    __shared__ h16 Bs[2][BN * SB];

    const int tid   = threadIdx.x;
    const int lane  = tid & 31;
    const int wave  = tid >> 5;
    const int waveM = wave & 3;    // 0..3  -> 32-row band
    const int waveN = wave >> 2;   // 0..1  -> 64-col band
    const int laneL = lane & 15;   // M-index for A frag, N-index for B frag / C col
    const int hi    = lane >> 4;   // half-selector

    const unsigned int asOff[2] = { (unsigned int)(uintptr_t)&As[0][0],
                                    (unsigned int)(uintptr_t)&As[1][0] };
    const unsigned int bsOff[2] = { (unsigned int)(uintptr_t)&Bs[0][0],
                                    (unsigned int)(uintptr_t)&Bs[1][0] };

    v8f acc[2][4];
    v8f vzero = {};
#pragma unroll
    for (int mt = 0; mt < 2; ++mt)
#pragma unroll
        for (int nt = 0; nt < 4; ++nt) acc[mt][nt] = vzero;

    const int nk = K / BK;
    // prologue: DMA first k-tile into buffer 0
    if (wave == 0)      tdm_load_tile_f16(asOff[0], A,  lda);
    else if (wave == 1) tdm_load_tile_f16(bsOff[0], Bt, ldb);

    for (int ikt = 0; ikt < nk; ++ikt) {
        const int cur = ikt & 1;
        // DMA next tile into the other buffer, then wait for current tile.
        if (wave == 0) {
            if (ikt + 1 < nk) {
                tdm_load_tile_f16(asOff[cur ^ 1], A + (long long)(ikt + 1) * BK, lda);
                __builtin_amdgcn_s_wait_tensorcnt(1);   // current A tile landed
            } else {
                __builtin_amdgcn_s_wait_tensorcnt(0);
            }
        } else if (wave == 1) {
            if (ikt + 1 < nk) {
                tdm_load_tile_f16(bsOff[cur ^ 1], Bt + (long long)(ikt + 1) * BK, ldb);
                __builtin_amdgcn_s_wait_tensorcnt(1);   // current B tile landed
            } else {
                __builtin_amdgcn_s_wait_tensorcnt(0);
            }
        }
        __syncthreads();   // tiles visible to all waves

        // A fragments: lane laneL = M; halves[0..7]=K[hi*8..], halves[8..15]=K[16+hi*8..]
        v16h afrag[2];
#pragma unroll
        for (int mt = 0; mt < 2; ++mt) {
            const int row = waveM * 32 + mt * 16 + laneL;
            v8h lo = *(const v8h*)&As[cur][row * SA + hi * 8];
            v8h hh = *(const v8h*)&As[cur][row * SA + 16 + hi * 8];
#pragma unroll
            for (int i = 0; i < 8; ++i) { afrag[mt][i] = lo[i]; afrag[mt][8 + i] = hh[i]; }
        }
        // B fragments: lane laneL = N; halves[0..7]=K[hi*8..], halves[8..15]=K[16+hi*8..]
        v16h bfrag[4];
#pragma unroll
        for (int nt = 0; nt < 4; ++nt) {
            const int col = waveN * 64 + nt * 16 + laneL;
            v8h lo = *(const v8h*)&Bs[cur][col * SB + hi * 8];
            v8h hh = *(const v8h*)&Bs[cur][col * SB + 16 + hi * 8];
#pragma unroll
            for (int i = 0; i < 8; ++i) { bfrag[nt][i] = lo[i]; bfrag[nt][8 + i] = hh[i]; }
        }
#pragma unroll
        for (int mt = 0; mt < 2; ++mt)
#pragma unroll
            for (int nt = 0; nt < 4; ++nt)
                acc[mt][nt] = __builtin_amdgcn_wmma_f32_16x16x32_f16(
                    false, afrag[mt], false, bfrag[nt],
                    (short)0, acc[mt][nt], false, false);

        __syncthreads();   // all LDS reads retired before buffer is re-DMA'd
    }

    // Epilogue. C/D layout: VGPR r -> M = r + 8*hi, N = laneL.
#pragma unroll
    for (int mt = 0; mt < 2; ++mt) {
#pragma unroll
        for (int nt = 0; nt < 4; ++nt) {
            const int gn = n0 + waveN * 64 + nt * 16 + laneL;
            const int gmb = m0 + waveM * 32 + mt * 16 + hi * 8;
            const float bcol = bias ? bias[gn] : 0.0f;
#pragma unroll
            for (int r = 0; r < 8; ++r) {
                const long long gm = gmb + r;
                float val = acc[mt][nt][r] + bcol;
                if (residual) val += residual[z * zCf + gm * ldcf + gn];
                if (relu)     val  = fmaxf(val, 0.0f);
                if (Cf) Cf[z * zCf  + gm * ldcf + gn] = val;
                if (Ch) Ch[z * zCh  + gm * ldch + gn] = (h16)val;
            }
        }
    }
}

// ---------------------------------------------------------------------------
// x[bt][c] = tok_emb[idx[bt]][c] + pos_emb[t][c]
// ---------------------------------------------------------------------------
__global__ __launch_bounds__(NTHREADS)
void embed_kernel(const int* __restrict__ idx, const float* __restrict__ tok,
                  const float* __restrict__ pos, float* __restrict__ x,
                  int T, int C)
{
    const int bt = blockIdx.x;
    const int t  = bt % T;
    const long long id = idx[bt];
    const float* te = tok + id * (long long)C;
    const float* pe = pos + (long long)t * C;
    float* xr = x + (long long)bt * C;
    for (int c = threadIdx.x; c < C; c += NTHREADS) xr[c] = te[c] + pe[c];
}

// ---------------------------------------------------------------------------
// LayerNorm over last dim (population variance, eps=1e-5), f16 output.
// ---------------------------------------------------------------------------
__global__ __launch_bounds__(NTHREADS)
void layernorm_f16(const float* __restrict__ x, const float* __restrict__ g,
                   const float* __restrict__ b, h16* __restrict__ out, int C)
{
    const int row = blockIdx.x;
    const float* xr = x + (long long)row * C;
    float s = 0.0f, s2 = 0.0f;
    for (int c = threadIdx.x; c < C; c += NTHREADS) {
        const float v = xr[c]; s += v; s2 += v * v;
    }
    __shared__ float rs[NTHREADS], rs2[NTHREADS];
    rs[threadIdx.x] = s; rs2[threadIdx.x] = s2;
    __syncthreads();
    for (int st = NTHREADS / 2; st > 0; st >>= 1) {
        if (threadIdx.x < st) { rs[threadIdx.x] += rs[threadIdx.x + st];
                                rs2[threadIdx.x] += rs2[threadIdx.x + st]; }
        __syncthreads();
    }
    const float mu  = rs[0] / (float)C;
    const float var = rs2[0] / (float)C - mu * mu;
    const float inv = rsqrtf(var + 1e-5f);
    h16* orow = out + (long long)row * C;
    for (int c = threadIdx.x; c < C; c += NTHREADS)
        orow[c] = (h16)((xr[c] - mu) * inv * g[c] + b[c]);
}

// ---------------------------------------------------------------------------
// Causal softmax row kernel: P[i][j] = softmax_j<=i(S[i][j]*scale), 0 above.
// ---------------------------------------------------------------------------
__global__ __launch_bounds__(NTHREADS)
void softmax_causal(const float* __restrict__ S, h16* __restrict__ P,
                    int T, float scale)
{
    const int i = blockIdx.x;
    const float* srow = S + (long long)i * T;
    h16* prow = P + (long long)i * T;
    const int n = i + 1;

    __shared__ float red[NTHREADS];
    float m = -1e30f;
    for (int j = threadIdx.x; j < n; j += NTHREADS) m = fmaxf(m, srow[j] * scale);
    red[threadIdx.x] = m; __syncthreads();
    for (int st = NTHREADS / 2; st > 0; st >>= 1) {
        if (threadIdx.x < st) red[threadIdx.x] = fmaxf(red[threadIdx.x], red[threadIdx.x + st]);
        __syncthreads();
    }
    const float mmax = red[0];
    __syncthreads();

    float sum = 0.0f;
    for (int j = threadIdx.x; j < n; j += NTHREADS) sum += __expf(srow[j] * scale - mmax);
    red[threadIdx.x] = sum; __syncthreads();
    for (int st = NTHREADS / 2; st > 0; st >>= 1) {
        if (threadIdx.x < st) red[threadIdx.x] += red[threadIdx.x + st];
        __syncthreads();
    }
    const float invs = 1.0f / red[0];

    for (int j = threadIdx.x; j < T; j += NTHREADS)
        prow[j] = (j < n) ? (h16)(__expf(srow[j] * scale - mmax) * invs) : (h16)0.0f;
}

// ---------------------------------------------------------------------------
// Transpose + convert: out[c][r] = (f16) in[r][c];  in is R x C f32.
// ---------------------------------------------------------------------------
__global__ __launch_bounds__(NTHREADS)
void transpose_cvt_f32f16(const float* __restrict__ in, h16* __restrict__ out,
                          int R, int C)
{
    __shared__ float tile[32][33];
    const int c0 = blockIdx.x * 32, r0 = blockIdx.y * 32;
    const int tx = threadIdx.x & 31, ty = threadIdx.x >> 5;   // 32x8
#pragma unroll
    for (int k = 0; k < 4; ++k) {
        const int r = r0 + ty + k * 8, c = c0 + tx;
        tile[ty + k * 8][tx] = (r < R && c < C) ? in[(long long)r * C + c] : 0.0f;
    }
    __syncthreads();
#pragma unroll
    for (int k = 0; k < 4; ++k) {
        const int rr = r0 + tx, cc = c0 + ty + k * 8;
        if (rr < R && cc < C)
            out[(long long)cc * R + rr] = (h16)tile[tx][ty + k * 8];
    }
}

// f16 -> f16 transpose (for V^T)
__global__ __launch_bounds__(NTHREADS)
void transpose_f16(const h16* __restrict__ in, h16* __restrict__ out,
                   int R, int C)
{
    __shared__ h16 tile[32][34];
    const int c0 = blockIdx.x * 32, r0 = blockIdx.y * 32;
    const int tx = threadIdx.x & 31, ty = threadIdx.x >> 5;
#pragma unroll
    for (int k = 0; k < 4; ++k) {
        const int r = r0 + ty + k * 8, c = c0 + tx;
        tile[ty + k * 8][tx] = (r < R && c < C) ? in[(long long)r * C + c] : (h16)0.0f;
    }
    __syncthreads();
#pragma unroll
    for (int k = 0; k < 4; ++k) {
        const int rr = r0 + tx, cc = c0 + ty + k * 8;
        if (rr < R && cc < C)
            out[(long long)cc * R + rr] = tile[tx][ty + k * 8];
    }
}

// ---------------------------------------------------------------------------
// Host driver
// ---------------------------------------------------------------------------
extern "C" void kernel_launch(void* const* d_in, const int* in_sizes, int n_in,
                              void* d_out, int out_size, void* d_ws, size_t ws_size,
                              hipStream_t stream)
{
    (void)in_sizes; (void)n_in; (void)out_size; (void)ws_size;
    constexpr int V = 32000, C = 1024, H = 8, T = 2048, B = 2, D = 1024, F = 4096;
    constexpr int BT = B * T;                 // 4096 rows through every GEMM
    const float scale = 0.03125f;             // D^-0.5 = 1/32

    // ---- inputs ----
    const int*   idx  = (const int*)  d_in[0];
    const float* tok  = (const float*)d_in[1];
    const float* pos  = (const float*)d_in[2];
    const float* Wq   = (const float*)d_in[3];
    const float* Wk   = (const float*)d_in[4];
    const float* Wv   = (const float*)d_in[5];
    const float* Wo   = (const float*)d_in[6];
    const float* bo   = (const float*)d_in[7];
    const float* W1   = (const float*)d_in[8];
    const float* b1   = (const float*)d_in[9];
    const float* W2   = (const float*)d_in[10];
    const float* b2   = (const float*)d_in[11];
    const float* g1   = (const float*)d_in[12];
    const float* be1  = (const float*)d_in[13];
    const float* g2   = (const float*)d_in[14];
    const float* be2  = (const float*)d_in[15];
    const float* gf   = (const float*)d_in[16];
    const float* bef  = (const float*)d_in[17];
    const float* Wlm  = (const float*)d_in[18];
    const float* blm  = (const float*)d_in[19];
    float* out = (float*)d_out;

    // ---- workspace layout (~483 MB) ----
    char*  base = (char*)d_ws;
    size_t off  = 0;
    auto alloc = [&](size_t bytes) -> char* {
        char* p = base + off;
        off = (off + bytes + 255) & ~(size_t)255;
        return p;
    };
    float* x     = (float*)alloc((size_t)BT * C * 4);          //  16 MB residual stream
    h16*   hf    = (h16*)  alloc((size_t)BT * C * 2);          //   8 MB LN output (reused 3x)
    h16*   WqT   = (h16*)  alloc((size_t)H * D * C * 2);       //  16 MB
    h16*   WkT   = (h16*)  alloc((size_t)H * D * C * 2);       //  16 MB
    h16*   WvT   = (h16*)  alloc((size_t)H * D * C * 2);       //  16 MB
    h16*   WoT   = (h16*)  alloc((size_t)C * H * D * 2);       //  16 MB  [C][H*D]
    h16*   W1T   = (h16*)  alloc((size_t)F * C * 2);           //   8 MB  [F][C]
    h16*   W2T   = (h16*)  alloc((size_t)C * F * 2);           //   8 MB  [C][F]
    h16*   WlmT  = (h16*)  alloc((size_t)V * C * 2);           //  64 MB  [V][C]
    h16*   q     = (h16*)  alloc((size_t)B * H * T * D * 2);   //  64 MB
    h16*   k     = (h16*)  alloc((size_t)B * H * T * D * 2);   //  64 MB
    h16*   v     = (h16*)  alloc((size_t)B * H * T * D * 2);   //  64 MB
    h16*   vT    = (h16*)  alloc((size_t)D * T * 2);           //   4 MB per-(b,h) scratch
    float* sc    = (float*)alloc((size_t)T * T * 4);           //  16 MB per-(b,h) scores
    h16*   P     = (h16*)  alloc((size_t)T * T * 2);           //   8 MB per-(b,h) softmax
    h16*   att   = (h16*)  alloc((size_t)BT * H * D * 2);      //  64 MB [B*T][H*D]
    h16*   ffn1  = (h16*)  alloc((size_t)BT * F * 2);          //  32 MB

    const dim3 blk(NTHREADS);

    // 1) embedding
    embed_kernel<<<BT, blk, 0, stream>>>(idx, tok, pos, x, T, C);

    // 2) transpose + f16-convert all weights (B operands stored [N][K])
    for (int h = 0; h < H; ++h) {
        transpose_cvt_f32f16<<<dim3(D / 32, C / 32), blk, 0, stream>>>(
            Wq + (size_t)h * C * D, WqT + (size_t)h * D * C, C, D);
        transpose_cvt_f32f16<<<dim3(D / 32, C / 32), blk, 0, stream>>>(
            Wk + (size_t)h * C * D, WkT + (size_t)h * D * C, C, D);
        transpose_cvt_f32f16<<<dim3(D / 32, C / 32), blk, 0, stream>>>(
            Wv + (size_t)h * C * D, WvT + (size_t)h * D * C, C, D);
    }
    transpose_cvt_f32f16<<<dim3(C / 32, (H * D) / 32), blk, 0, stream>>>(Wo, WoT, H * D, C);
    transpose_cvt_f32f16<<<dim3(F / 32, C / 32), blk, 0, stream>>>(W1, W1T, C, F);
    transpose_cvt_f32f16<<<dim3(C / 32, F / 32), blk, 0, stream>>>(W2, W2T, F, C);
    transpose_cvt_f32f16<<<dim3(V / 32, C / 32), blk, 0, stream>>>(Wlm, WlmT, C, V);

    // 3) LN1 -> hf (f16)
    layernorm_f16<<<BT, blk, 0, stream>>>(x, g1, be1, hf, C);

    // 4) Q/K/V projections per head, batched over B via grid.z
    for (int h = 0; h < H; ++h) {
        const long long zA = (long long)T * C;        // hf batch stride
        const long long zC = (long long)H * T * D;    // q[b][h] batch stride
        gemm_wmma_f16<<<dim3(D / BN, T / BM, B), blk, 0, stream>>>(
            hf, C, zA, WqT + (size_t)h * D * C, C, 0,
            nullptr, 0, 0, q + (size_t)h * T * D, D, zC,
            nullptr, nullptr, C, 0, 0);
        gemm_wmma_f16<<<dim3(D / BN, T / BM, B), blk, 0, stream>>>(
            hf, C, zA, WkT + (size_t)h * D * C, C, 0,
            nullptr, 0, 0, k + (size_t)h * T * D, D, zC,
            nullptr, nullptr, C, 0, 0);
        gemm_wmma_f16<<<dim3(D / BN, T / BM, B), blk, 0, stream>>>(
            hf, C, zA, WvT + (size_t)h * D * C, C, 0,
            nullptr, 0, 0, v + (size_t)h * T * D, D, zC,
            nullptr, nullptr, C, 0, 0);
    }

    // 5) attention per (b,h): scores (causal-skipped) -> softmax -> P @ V
    for (int b = 0; b < B; ++b) {
        for (int h = 0; h < H; ++h) {
            const h16* qbh = q + ((size_t)(b * H + h)) * T * D;
            const h16* kbh = k + ((size_t)(b * H + h)) * T * D;
            const h16* vbh = v + ((size_t)(b * H + h)) * T * D;
            // scores = q @ k^T  (Bt = k directly, [s][d])
            gemm_wmma_f16<<<dim3(T / BN, T / BM, 1), blk, 0, stream>>>(
                qbh, D, 0, kbh, D, 0,
                sc, T, 0, nullptr, 0, 0,
                nullptr, nullptr, D, 0, /*causal=*/1);
            softmax_causal<<<T, blk, 0, stream>>>(sc, P, T, scale);
            // V^T for the P@V GEMM B operand
            transpose_f16<<<dim3(D / 32, T / 32), blk, 0, stream>>>(vbh, vT, T, D);
            // att[b, :, h*D:(h+1)*D] = P @ V   (f16 out, row stride H*D)
            gemm_wmma_f16<<<dim3(D / BN, T / BM, 1), blk, 0, stream>>>(
                P, T, 0, vT, T, 0,
                nullptr, 0, 0,
                att + (size_t)b * T * H * D + (size_t)h * D, H * D, 0,
                nullptr, nullptr, T, 0, 0);
        }
    }

    // 6) x += att @ Wo + bo
    gemm_wmma_f16<<<dim3(C / BN, BT / BM, 1), blk, 0, stream>>>(
        att, H * D, 0, WoT, H * D, 0,
        x, C, 0, nullptr, 0, 0,
        bo, x, H * D, 0, 0);

    // 7) FFN: ffn1 = relu(ln2(x) @ W1 + b1); x += ffn1 @ W2 + b2
    layernorm_f16<<<BT, blk, 0, stream>>>(x, g2, be2, hf, C);
    gemm_wmma_f16<<<dim3(F / BN, BT / BM, 1), blk, 0, stream>>>(
        hf, C, 0, W1T, C, 0,
        nullptr, 0, 0, ffn1, F, 0,
        b1, nullptr, C, /*relu=*/1, 0);
    gemm_wmma_f16<<<dim3(C / BN, BT / BM, 1), blk, 0, stream>>>(
        ffn1, F, 0, W2T, F, 0,
        x, C, 0, nullptr, 0, 0,
        b2, x, F, 0, 0);

    // 8) logits = lnf(x) @ Wlm + blm  -> d_out (f32, [B*T][V])
    layernorm_f16<<<BT, blk, 0, stream>>>(x, gf, bef, hf, C);
    gemm_wmma_f16<<<dim3(V / BN, BT / BM, 1), blk, 0, stream>>>(
        hf, C, 0, WlmT, C, 0,
        out, V, 0, nullptr, 0, 0,
        blm, nullptr, C, 0, 0);
}
